// STGCN_Model_7198365188832
// MI455X (gfx1250) — matrix operand order repaired and checked by reading, not compiled, not run on hardware
//
#include <hip/hip_runtime.h>
#include <hip/hip_bf16.h>

// ST-GCN for MI455X (gfx1250, wave32, WMMA, async-to-LDS staging).
// Pipeline:
//   prep:    build sparse normalized adjacency (53 nnz) + f16-transposed weights
//   gcn1..3: per 16-row tile: async-copy raw tile to LDS (blocks 2/3), sparse
//            N-mix into f16 A-tile in LDS, WMMA f16->f32 GEMM vs Wt, bias+relu,
//            emit f16 H (blocks 1,2) + per-block BN partials (no atomics)
//   bnstat:  reduce partials -> per-channel scale/shift (BN folded to affine)
//   tail:    fold scale3/shift3 into block-3 linear partials, V/T mean pool,
//            256->60 classifier.

typedef _Float16 half_t;
typedef _Float16 v16h __attribute__((ext_vector_type(16)));
typedef float    v8f  __attribute__((ext_vector_type(8)));
typedef unsigned int u32x4 __attribute__((ext_vector_type(4)));

// async-to-LDS builtin takes int4* pointers (global src, LDS dst)
typedef int v4i_g __attribute__((vector_size(16)));
typedef __attribute__((address_space(1))) v4i_g* as1_v4ip;
typedef __attribute__((address_space(3))) v4i_g* as3_v4ip;

#define BTOT   16384   // B*T
#define VN     17

__device__ __forceinline__ v16h frag16(const half_t* p0, const half_t* p1) {
    union { v16h h; u32x4 u[2]; } c;
    c.u[0] = *(const u32x4*)p0;
    c.u[1] = *(const u32x4*)p1;
    return c.h;
}

// 16-byte async global->LDS copy (gfx1250 GLOBAL_LOAD_ASYNC_TO_LDS_B128)
__device__ __forceinline__ void async_copy16(const void* g, void* l) {
#if defined(__has_builtin) && __has_builtin(__builtin_amdgcn_global_load_async_to_lds_b128)
    __builtin_amdgcn_global_load_async_to_lds_b128(
        (as1_v4ip)(v4i_g*)g, (as3_v4ip)(v4i_g*)l, 0, 0);
#else
    unsigned lo = (unsigned)(uintptr_t)l;   // flat->LDS addr truncation
    asm volatile("global_load_async_to_lds_b128 %0, %1, off"
                 :: "v"(lo), "v"(g) : "memory");
#endif
}

__device__ __forceinline__ void wait_async0() {
#if defined(__has_builtin) && __has_builtin(__builtin_amdgcn_s_wait_asynccnt)
    __builtin_amdgcn_s_wait_asynccnt(0);
#else
    asm volatile("s_wait_asynccnt 0x0" ::: "memory");
#endif
}

// ---------------------------------------------------------------------------
// prep: adjacency -> sparse normalized N (nnz list), W -> f16 transposed+padded
// ---------------------------------------------------------------------------
__global__ void stgcn_prep(const int* __restrict__ ei,
                           const float* __restrict__ W1, const float* __restrict__ W2,
                           const float* __restrict__ W3,
                           int* __restrict__ nnzoff, int* __restrict__ nnzu,
                           float* __restrict__ nnzw,
                           half_t* __restrict__ Wt1, half_t* __restrict__ Wt2,
                           half_t* __restrict__ Wt3) {
    int gtid = blockIdx.x * blockDim.x + threadIdx.x;
    int gsz  = gridDim.x * blockDim.x;
    if (gtid == 0) {
        float A[VN][VN];
        for (int v = 0; v < VN; v++)
            for (int u = 0; u < VN; u++) A[v][u] = 0.f;
        for (int e = 0; e < 36; e++) { int s = ei[e], d = ei[36 + e]; A[d][s] += 1.f; }
        for (int v = 0; v < VN; v++) A[v][v] += 1.f;
        float dinv[VN];
        for (int v = 0; v < VN; v++) {
            float s = 0.f;
            for (int u = 0; u < VN; u++) s += A[v][u];
            dinv[v] = rsqrtf(s);
        }
        int cnt = 0;
        for (int v = 0; v < VN; v++) {
            nnzoff[v] = cnt;
            for (int u = 0; u < VN; u++) {
                if (A[v][u] != 0.f) {
                    nnzu[cnt] = u;
                    nnzw[cnt] = dinv[v] * A[v][u] * dinv[u];
                    cnt++;
                }
            }
        }
        nnzoff[VN] = cnt;
    }
    // Wt[cout][k] = W[k][cout], zero-padded in k
    for (int i = gtid; i < 64 * 32; i += gsz) {
        int c = i / 32, k = i % 32;
        Wt1[i] = (half_t)(k < 3 ? W1[k * 64 + c] : 0.f);
    }
    for (int i = gtid; i < 128 * 64; i += gsz) {
        int c = i / 64, k = i % 64;
        Wt2[i] = (half_t)W2[k * 128 + c];
    }
    for (int i = gtid; i < 256 * 128; i += gsz) {
        int c = i / 128, k = i % 128;
        Wt3[i] = (half_t)W3[k * 256 + c];
    }
}

// ---------------------------------------------------------------------------
// GCN block kernel: one 16-row (b,t) tile per workgroup.
//   Xraw[m][u][k]  <- async bulk copy of contiguous input tile  (blocks 2,3)
//   Xm[v][m][k]    = sum_u N[v,u] * affine(Xraw[m][u][k])        (LDS, f16)
//   H[m][v][cout]  = relu( WMMA(Xm[v], Wt) + bias )
// Emits per-block per-channel sum/sumsq partials (plain stores, no atomics).
// ---------------------------------------------------------------------------
template<int CIN, int CINPAD, int COUT, int NWAVES, bool FIRST, bool STORE_H>
__global__ __launch_bounds__(NWAVES * 32)
void stgcn_gcn(const void* __restrict__ in_,
               const float* __restrict__ scale_in, const float* __restrict__ shift_in,
               const half_t* __restrict__ Wt, const float* __restrict__ bias,
               const int* __restrict__ nnzoff, const int* __restrict__ nnzu,
               const float* __restrict__ nnzw,
               half_t* __restrict__ Hout,
               float* __restrict__ Psum, float* __restrict__ Psq) {
    constexpr int KC  = CINPAD / 32;   // k-chunks of 32
    constexpr int NT  = COUT / 16;     // 16-wide n tiles
    constexpr int NCH = VN * COUT;
    const int tid  = threadIdx.x;
    const int nthr = NWAVES * 32;
    const int blk  = blockIdx.x;
    const long bt0 = (long)blk * 16;

    __shared__ int   soff[VN + 1];
    __shared__ int   snzu[64];
    __shared__ float snzw[64];
    extern __shared__ __align__(16) char smem_raw[];
    half_t* Xm   = (half_t*)smem_raw;            // [VN][16][CINPAD] A-tile
    half_t* Xraw = Xm + VN * 16 * CINPAD;        // [16][VN][CIN] raw tile (!FIRST)

    if (tid < VN + 1) soff[tid] = nnzoff[tid];
    if (tid < 64) { snzu[tid] = nnzu[tid]; snzw[tid] = nnzw[tid]; }

    if constexpr (!FIRST) {
        // contiguous tile: 16 rows x VN x CIN halves
        constexpr int TILEBYTES = 16 * VN * CIN * 2;
        const char* gsrc = (const char*)in_ + (size_t)bt0 * (VN * CIN) * sizeof(half_t);
        char* ldst = (char*)Xraw;
        for (int c = tid * 16; c < TILEBYTES; c += nthr * 16)
            async_copy16(gsrc + c, ldst + c);
        wait_async0();
    }
    __syncthreads();

    const float* xin = (const float*)in_;

    // sparse adjacency mix + BN-affine of previous block -> f16 A-tile
    for (int i = tid; i < VN * 16 * CINPAD; i += nthr) {
        int v = i / (16 * CINPAD);
        int r = i - v * (16 * CINPAD);
        int m = r / CINPAD;
        int k = r - m * CINPAD;
        float a = 0.f;
        if (k < CIN) {
            int e0 = soff[v], e1 = soff[v + 1];
            for (int j = e0; j < e1; ++j) {
                int u = snzu[j];
                float w = snzw[j];
                float xv;
                if constexpr (FIRST) {
                    xv = xin[((bt0 + m) * VN + u) * CIN + k];
                } else {
                    xv = (float)Xraw[(m * VN + u) * CIN + k];
                    int ch = u * CIN + k;
                    xv = xv * scale_in[ch] + shift_in[ch];
                }
                a += w * xv;
            }
        }
        Xm[i] = (half_t)a;
    }
    __syncthreads();

    const int wv = tid >> 5, lane = tid & 31;
    const int lm = lane & 15, hi = lane >> 4;

    for (int nt = wv; nt < NT; nt += NWAVES) {
        const int cout0 = nt * 16;
        const int col   = cout0 + lm;
        // B fragments: Wt row-major [cout][CINPAD]; lane n holds col n,
        // K = hi*16 .. hi*16+15 of the 32-chunk
        v16h bfr[KC];
        for (int kc = 0; kc < KC; kc++) {
            size_t bb = (size_t)col * CINPAD + kc * 32 + hi * 16;
            bfr[kc] = frag16(Wt + bb, Wt + bb + 8);
        }
        const float bcol = bias[col];

        for (int v = 0; v < VN; ++v) {
            v8f acc = {};
            for (int kc = 0; kc < KC; kc++) {
                // A fragment: lane m holds row m; K = hi*16 .. hi*16+15
                int ab = (v * 16 + lm) * CINPAD + kc * 32 + hi * 16;
                v16h af = frag16(Xm + ab, Xm + ab + 8);
                acc = __builtin_amdgcn_wmma_f32_16x16x32_f16(
                        false, af, false, bfr[kc], (short)0, acc, false, false);
            }
            float s = 0.f, q = 0.f;
            float vals[8];
            for (int r = 0; r < 8; r++) {
                float x = acc[r] + bcol;       // C: lane<16 rows 0..7, lane>=16 rows 8..15
                x = fmaxf(x, 0.f);
                vals[r] = x;
                s += x;
                q += x * x;
            }
            if constexpr (STORE_H) {
                for (int r = 0; r < 8; r++) {
                    Hout[((size_t)(bt0 + hi * 8 + r) * VN + v) * COUT + col] =
                        (half_t)vals[r];
                }
            }
            s += __shfl_xor(s, 16);
            q += __shfl_xor(q, 16);
            if (lane < 16) {
                size_t pb = (size_t)blk * NCH + (size_t)v * COUT + col;
                Psum[pb] = s;
                Psq[pb]  = q;
            }
        }
    }
}

// ---------------------------------------------------------------------------
// BN stats: reduce 1024 per-block partials -> affine scale/shift per channel
// ---------------------------------------------------------------------------
__global__ void stgcn_bnstat(const float* __restrict__ Psum, const float* __restrict__ Psq,
                             const float* __restrict__ g, const float* __restrict__ be,
                             float* __restrict__ scale, float* __restrict__ shift,
                             int nch, int nblk) {
    int ch = blockIdx.x * blockDim.x + threadIdx.x;
    if (ch >= nch) return;
    float s = 0.f, q = 0.f;
    for (int b = 0; b < nblk; b++) {
        s += Psum[(size_t)b * nch + ch];
        q += Psq[(size_t)b * nch + ch];
    }
    const float invM = 1.f / (float)BTOT;
    float mean = s * invM;
    float var  = q * invM - mean * mean;
    float inv  = rsqrtf(var + 1e-5f);
    float sc   = g[ch] * inv;
    scale[ch] = sc;
    shift[ch] = be[ch] - mean * sc;
}

// ---------------------------------------------------------------------------
// tail: fold BN3 affine into block-3 linear sums, mean over V,T, classifier
// pooled[b,c] = ( sum_v scale3[v,c]*S3[b,v,c] + T*sum_v shift3[v,c] ) / (T*V)
// out[b,k]    = bf[k] + sum_c pooled[b,c]*Wf[c,k]
// ---------------------------------------------------------------------------
__global__ void stgcn_tail(const float* __restrict__ P3sum,
                           const float* __restrict__ scale3, const float* __restrict__ shift3,
                           const float* __restrict__ Wf, const float* __restrict__ bfc,
                           float* __restrict__ out) {
    int b = blockIdx.x;
    int c = threadIdx.x;               // 256 threads = 256 channels
    __shared__ float pl[256];
    float acc = 0.f;
    for (int v = 0; v < VN; v++) {
        int ch = v * 256 + c;
        float s3 = 0.f;
        for (int j = 0; j < 32; j++)   // 32 row-tiles per batch element
            s3 += P3sum[(size_t)(b * 32 + j) * (VN * 256) + ch];
        acc += scale3[ch] * s3 + 512.f * shift3[ch];
    }
    pl[c] = acc * (1.f / (512.f * (float)VN));
    __syncthreads();
    if (c < 60) {
        float o = bfc[c];
        for (int cc = 0; cc < 256; cc++) o += pl[cc] * Wf[cc * 60 + c];
        out[b * 60 + c] = o;
    }
}

// ---------------------------------------------------------------------------
extern "C" void kernel_launch(void* const* d_in, const int* in_sizes, int n_in,
                              void* d_out, int out_size, void* d_ws, size_t ws_size,
                              hipStream_t stream) {
    const float* x   = (const float*)d_in[0];
    const int*   ei  = (const int*)d_in[1];
    const float* W1  = (const float*)d_in[2];
    const float* b1  = (const float*)d_in[3];
    const float* g1  = (const float*)d_in[4];
    const float* be1 = (const float*)d_in[5];
    const float* W2  = (const float*)d_in[6];
    const float* b2  = (const float*)d_in[7];
    const float* g2  = (const float*)d_in[8];
    const float* be2 = (const float*)d_in[9];
    const float* W3  = (const float*)d_in[10];
    const float* b3  = (const float*)d_in[11];
    const float* g3  = (const float*)d_in[12];
    const float* be3 = (const float*)d_in[13];
    const float* Wf  = (const float*)d_in[14];
    const float* bfc = (const float*)d_in[15];
    float* out = (float*)d_out;

    size_t off = 0;
    auto carve = [&](size_t bytes) -> char* {
        char* p = (char*)d_ws + off;
        off += (bytes + 255) & ~(size_t)255;
        return p;
    };
    int*    nnzoff = (int*)carve((VN + 1) * sizeof(int));
    int*    nnzu   = (int*)carve(289 * sizeof(int));
    float*  nnzw   = (float*)carve(289 * sizeof(float));
    half_t* Wt1 = (half_t*)carve((size_t)64 * 32 * 2);
    half_t* Wt2 = (half_t*)carve((size_t)128 * 64 * 2);
    half_t* Wt3 = (half_t*)carve((size_t)256 * 128 * 2);
    float* P1s = (float*)carve(1024UL * (VN * 64) * 4);
    float* P1q = (float*)carve(1024UL * (VN * 64) * 4);
    float* P2s = (float*)carve(1024UL * (VN * 128) * 4);
    float* P2q = (float*)carve(1024UL * (VN * 128) * 4);
    float* P3s = (float*)carve(1024UL * (VN * 256) * 4);
    float* P3q = (float*)carve(1024UL * (VN * 256) * 4);
    float* scale1 = (float*)carve((VN * 64) * 4);
    float* shift1 = (float*)carve((VN * 64) * 4);
    float* scale2 = (float*)carve((VN * 128) * 4);
    float* shift2 = (float*)carve((VN * 128) * 4);
    float* scale3 = (float*)carve((VN * 256) * 4);
    float* shift3 = (float*)carve((VN * 256) * 4);
    half_t* H1 = (half_t*)carve((size_t)BTOT * VN * 64 * 2);
    half_t* H2 = (half_t*)carve((size_t)BTOT * VN * 128 * 2);
    (void)ws_size; (void)in_sizes; (void)n_in; (void)out_size;

    // LDS budgets: Xm + (blocks 2/3) raw tile
    const int lds1 = VN * 16 * 32 * 2;
    const int lds2 = VN * 16 * 64 * 2 + 16 * VN * 64 * 2;
    const int lds3 = VN * 16 * 128 * 2 + 16 * VN * 128 * 2;
    (void)hipFuncSetAttribute(
        reinterpret_cast<const void*>(&stgcn_gcn<64, 64, 128, 8, false, true>),
        hipFuncAttributeMaxDynamicSharedMemorySize, lds2);
    (void)hipFuncSetAttribute(
        reinterpret_cast<const void*>(&stgcn_gcn<128, 128, 256, 8, false, false>),
        hipFuncAttributeMaxDynamicSharedMemorySize, lds3);

    stgcn_prep<<<64, 256, 0, stream>>>(ei, W1, W2, W3, nnzoff, nnzu, nnzw,
                                       Wt1, Wt2, Wt3);

    stgcn_gcn<3, 32, 64, 4, true, true>
        <<<BTOT / 16, 128, lds1, stream>>>(
            x, nullptr, nullptr, Wt1, b1, nnzoff, nnzu, nnzw, H1, P1s, P1q);
    stgcn_bnstat<<<(VN * 64 + 255) / 256, 256, 0, stream>>>(
        P1s, P1q, g1, be1, scale1, shift1, VN * 64, 1024);

    stgcn_gcn<64, 64, 128, 8, false, true>
        <<<BTOT / 16, 256, lds2, stream>>>(
            H1, scale1, shift1, Wt2, b2, nnzoff, nnzu, nnzw, H2, P2s, P2q);
    stgcn_bnstat<<<(VN * 128 + 255) / 256, 256, 0, stream>>>(
        P2s, P2q, g2, be2, scale2, shift2, VN * 128, 1024);

    stgcn_gcn<128, 128, 256, 8, false, false>
        <<<BTOT / 16, 256, lds3, stream>>>(
            H2, scale2, shift2, Wt3, b3, nnzoff, nnzu, nnzw, nullptr, P3s, P3q);
    stgcn_bnstat<<<(VN * 256 + 255) / 256, 256, 0, stream>>>(
        P3s, P3q, g3, be3, scale3, shift3, VN * 256, 1024);

    stgcn_tail<<<32, 256, 0, stream>>>(P3s, scale3, shift3, Wf, bfc, out);
}
